// BertAttentionEx_82789789598241
// MI455X (gfx1250) — compile-verified, matrix operand order checked by reading, not run on hardware
//
#include <hip/hip_runtime.h>

#define B_DIM 4
#define S_DIM 1024
#define HID 1024
#define NH 16
#define HD 64

typedef __bf16 bf16_t;
typedef __attribute__((ext_vector_type(8)))  __bf16 v8bf;
typedef __attribute__((ext_vector_type(16))) __bf16 v16bf;
typedef __attribute__((ext_vector_type(8)))  float  v8f;

__device__ __forceinline__ bf16_t f2bf(float f) {
  unsigned u = __builtin_bit_cast(unsigned, f);
  unsigned r = u + 0x7FFFu + ((u >> 16) & 1u);           // round-to-nearest-even
  return __builtin_bit_cast(bf16_t, (unsigned short)(r >> 16));
}

__device__ __forceinline__ v8f vzero8() {
  v8f z;
#pragma unroll
  for (int i = 0; i < 8; ++i) z[i] = 0.0f;
  return z;
}

// A/B fragment loader for V_WMMA_*_16X16X32_BF16 (wave32).
// ISA layout: lanes 0-15 hold row M=lane, K = {0..7, 16..23};
//             lanes 16-31 hold row M=lane-16, K = {8..15, 24..31}.
// B operand is fed from an N-major (i.e. B^T row-major) tile, same packing.
__device__ __forceinline__ v16bf ld_frag(const bf16_t* __restrict__ tile, int ld,
                                         int row, int k0, int khalf) {
  const bf16_t* p = tile + row * ld + k0 + (khalf << 3);
  v8bf lo = *(const v8bf*)p;          // K 0..7   (or 8..15)
  v8bf hi = *(const v8bf*)(p + 16);   // K 16..23 (or 24..31)
  return __builtin_shufflevector(lo, hi, 0, 1, 2, 3, 4, 5, 6, 7,
                                 8, 9, 10, 11, 12, 13, 14, 15);
}

__device__ __forceinline__ v8f wmma_bf16(v16bf a, v16bf b, v8f c) {
  return __builtin_amdgcn_wmma_f32_16x16x32_bf16(false, a, false, b, (short)0, c,
                                                 false, false);
}

// ---------------------------------------------------------------- fp32 -> bf16
__global__ __launch_bounds__(256) void cvt_bf16_kernel(const float* __restrict__ in,
                                                       bf16_t* __restrict__ out,
                                                       int n_in, int n_out) {
  for (int i = blockIdx.x * blockDim.x + threadIdx.x; i < n_out;
       i += gridDim.x * blockDim.x)
    out[i] = f2bf(i < n_in ? in[i] : 0.0f);
}

// ------------------------------------------------------- QKV projection GEMM
// C[4096,3072] = X[4096,1024] @ Wqkv^T, +bias, scattered to head-major Q/K/V.
// 64x64 tile, k-chunk 64; B fragments batched -> one dscnt wait per 4 WMMAs.
__global__ __launch_bounds__(128) void qkv_gemm_kernel(
    const bf16_t* __restrict__ X, const bf16_t* __restrict__ W,
    const float* __restrict__ bq, const float* __restrict__ bk,
    const float* __restrict__ bv, bf16_t* __restrict__ Qb,
    bf16_t* __restrict__ Kb, bf16_t* __restrict__ Vb) {
  __shared__ bf16_t sX[64 * 64];
  __shared__ bf16_t sW[64 * 64];
  const int n0 = blockIdx.x * 64;  // output feature tile [0,3072)
  const int m0 = blockIdx.y * 64;  // token tile [0,4096)
  const int tid = threadIdx.x;
  const int wave = tid >> 5, lane = tid & 31;
  const int lr = lane & 15, kh = lane >> 4, mb = kh * 8;

  v8f acc[4];
#pragma unroll
  for (int j = 0; j < 4; ++j) acc[j] = vzero8();

#pragma unroll 1
  for (int k0 = 0; k0 < HID; k0 += 64) {
    __syncthreads();
#pragma unroll
    for (int it = 0; it < 4; ++it) {
      int idx = (it * 128 + tid) * 8;
      int r = idx >> 6, c = idx & 63;
      *(v8bf*)(sX + idx) = *(const v8bf*)(X + (size_t)(m0 + r) * HID + k0 + c);
      *(v8bf*)(sW + idx) = *(const v8bf*)(W + (size_t)(n0 + r) * HID + k0 + c);
    }
    __syncthreads();
    v16bf a0 = ld_frag(sX, 64, wave * 16 + lr, 0, kh);
    v16bf a1 = ld_frag(sX, 64, wave * 16 + lr, 32, kh);
    v16bf bfr[4];
#pragma unroll
    for (int j = 0; j < 4; ++j) bfr[j] = ld_frag(sW, 64, j * 16 + lr, 0, kh);
#pragma unroll
    for (int j = 0; j < 4; ++j) acc[j] = wmma_bf16(a0, bfr[j], acc[j]);
#pragma unroll
    for (int j = 0; j < 4; ++j) bfr[j] = ld_frag(sW, 64, j * 16 + lr, 32, kh);
#pragma unroll
    for (int j = 0; j < 4; ++j) acc[j] = wmma_bf16(a1, bfr[j], acc[j]);
  }

  const int mat = n0 >> 10;        // 0=Q 1=K 2=V
  const int hc = n0 & 1023;        // head-base column within the matrix
  const float* bias = (mat == 0) ? bq : (mat == 1) ? bk : bv;
  bf16_t* Out = (mat == 0) ? Qb : (mat == 1) ? Kb : Vb;
  const int bb = m0 >> 10, l0 = m0 & 1023, hh = hc >> 6;
  bf16_t* base = Out + (((size_t)bb * NH + hh) * S_DIM) * HD;
#pragma unroll
  for (int j = 0; j < 4; ++j) {
    int d = j * 16 + lr;
    float bsv = bias[hc + d];
#pragma unroll
    for (int v = 0; v < 8; ++v) {
      int l = l0 + wave * 16 + mb + v;
      base[(size_t)l * HD + d] = f2bf(acc[j][v] + bsv);
    }
  }
}

// -------------------------------------------- fused flash attention + rel bias
// Block = (b, h, 64-row tile). 4 waves, each owns a 16-row strip.
__global__ __launch_bounds__(128) void attn_kernel(
    const bf16_t* __restrict__ Qb, const bf16_t* __restrict__ Kb,
    const bf16_t* __restrict__ Vb, const bf16_t* __restrict__ Eb,
    const float* __restrict__ mask, bf16_t* __restrict__ Ctx) {
  extern __shared__ char smem[];
  bf16_t* sQ  = (bf16_t*)smem;              // 64x64 bf16   8K
  bf16_t* sK  = sQ + 64 * 64;               // 64x64 bf16   8K
  bf16_t* sVt = sK + 64 * 64;               // V^T 64x64    8K
  bf16_t* sE  = sVt + 64 * 64;              // 128x64 bf16 16K
  float*  sQe = (float*)(sE + 128 * 64);    // 64x128 f32  32K
  float*  sKe = sQe + 64 * 128;             // 64x128 f32  32K
  bf16_t* sP  = (bf16_t*)(sKe + 64 * 128);  // 4 x 16x64    8K   => 112K total

  const int l0 = blockIdx.x * 64;
  const int h = blockIdx.y, b = blockIdx.z;
  const int bh = b * NH + h;
  const bf16_t* Qg = Qb + ((size_t)bh * S_DIM) * HD;
  const bf16_t* Kg = Kb + ((size_t)bh * S_DIM) * HD;
  const bf16_t* Vg = Vb + ((size_t)bh * S_DIM) * HD;
  const float* mrow = mask + (size_t)b * S_DIM;

  const int tid = threadIdx.x;
  const int wave = tid >> 5, lane = tid & 31;
  const int lr = lane & 15, kh = lane >> 4, mb = kh * 8;

#pragma unroll
  for (int it = 0; it < 4; ++it) {   // Q tile, flat copy
    int idx = (it * 128 + tid) * 8;
    *(v8bf*)(sQ + idx) = *(const v8bf*)(Qg + (size_t)l0 * HD + idx);
  }

  v8f ctx[4];
#pragma unroll
  for (int j = 0; j < 4; ++j) ctx[j] = vzero8();
  float mrun[8], lrun[8];
#pragma unroll
  for (int v = 0; v < 8; ++v) { mrun[v] = -1.0e30f; lrun[v] = 0.0f; }

  bf16_t* Pw = sP + wave * 16 * 64;

#pragma unroll 1
  for (int r0 = 0; r0 < S_DIM; r0 += 64) {
    __syncthreads();
#pragma unroll
    for (int it = 0; it < 4; ++it) {       // K row-major, V transposed
      int idx = (it * 128 + tid) * 8;
      *(v8bf*)(sK + idx) = *(const v8bf*)(Kg + (size_t)r0 * HD + idx);
      int r = idx >> 6, c = idx & 63;
      v8bf vv = *(const v8bf*)(Vg + (size_t)(r0 + r) * HD + c);
#pragma unroll
      for (int e = 0; e < 8; ++e) sVt[(c + e) * 64 + r] = vv[e];
    }
    const int ebase = l0 - r0 + 960;       // = l0-r0-63+1023; band of 128 rows
#pragma unroll
    for (int it = 0; it < 8; ++it) {
      int idx = (it * 128 + tid) * 8;
      *(v8bf*)(sE + idx) = *(const v8bf*)(Eb + (size_t)ebase * HD + idx);
    }
    if (r0 + 64 < S_DIM)
      __builtin_prefetch(Kg + (size_t)(r0 + 64) * HD + tid * 32, 0, 1);
    __syncthreads();

    // A fragments for this wave's Q-rows / K-rows, loaded once per tile
    v16bf aQ0 = ld_frag(sQ, 64, wave * 16 + lr, 0, kh);
    v16bf aQ1 = ld_frag(sQ, 64, wave * 16 + lr, 32, kh);
    v16bf aK0 = ld_frag(sK, 64, wave * 16 + lr, 0, kh);
    v16bf aK1 = ld_frag(sK, 64, wave * 16 + lr, 32, kh);

    // raw scores S = Q K^T (16x64 per wave); B frags batched per k-step
    v8f sc[4];
#pragma unroll
    for (int j = 0; j < 4; ++j) sc[j] = vzero8();
    {
      v16bf bfr[4];
#pragma unroll
      for (int j = 0; j < 4; ++j) bfr[j] = ld_frag(sK, 64, j * 16 + lr, 0, kh);
#pragma unroll
      for (int j = 0; j < 4; ++j) sc[j] = wmma_bf16(aQ0, bfr[j], sc[j]);
#pragma unroll
      for (int j = 0; j < 4; ++j) bfr[j] = ld_frag(sK, 64, j * 16 + lr, 32, kh);
#pragma unroll
      for (int j = 0; j < 4; ++j) sc[j] = wmma_bf16(aQ1, bfr[j], sc[j]);
    }

    // Qe = Q E^T, Ke = K E^T for this wave's 16 rows (cols u=0..127)
#pragma unroll
    for (int ju = 0; ju < 8; ++ju) {
      v16bf e0 = ld_frag(sE, 64, ju * 16 + lr, 0, kh);
      v16bf e1 = ld_frag(sE, 64, ju * 16 + lr, 32, kh);
      v8f aq = wmma_bf16(aQ1, e1, wmma_bf16(aQ0, e0, vzero8()));
      v8f ak = wmma_bf16(aK1, e1, wmma_bf16(aK0, e0, vzero8()));
#pragma unroll
      for (int v = 0; v < 8; ++v) {
        sQe[(wave * 16 + mb + v) * 128 + ju * 16 + lr] = aq[v];
        sKe[(wave * 16 + mb + v) * 128 + ju * 16 + lr] = ak[v];
      }
    }
    __syncthreads();

    // scores = (S + Qe[l][u] + Ke[r][u]) / 8 + mask ; online softmax.
    // Fold score values back into sc[] (dead after this point).
    float rmax[8];
#pragma unroll
    for (int v = 0; v < 8; ++v) rmax[v] = -1.0e30f;
#pragma unroll
    for (int j = 0; j < 4; ++j) {
      const int rj = j * 16 + lr;
      const float mval = mrow[r0 + rj];
#pragma unroll
      for (int v = 0; v < 8; ++v) {
        const int li = wave * 16 + mb + v;
        const int u = li - rj + 63;        // in [0,126]
        float s = (sc[j][v] + sQe[li * 128 + u] + sKe[rj * 128 + u]) * 0.125f + mval;
        sc[j][v] = s;
        rmax[v] = fmaxf(rmax[v], s);
      }
    }
#pragma unroll
    for (int off = 8; off >= 1; off >>= 1)
#pragma unroll
      for (int v = 0; v < 8; ++v)
        rmax[v] = fmaxf(rmax[v], __shfl_xor(rmax[v], off, 32));

    float scl[8], psum[8];
#pragma unroll
    for (int v = 0; v < 8; ++v) {
      float mnew = fmaxf(mrun[v], rmax[v]);
      scl[v] = __expf(mrun[v] - mnew);
      mrun[v] = mnew;
      psum[v] = 0.0f;
    }
#pragma unroll
    for (int j = 0; j < 4; ++j)
#pragma unroll
      for (int v = 0; v < 8; ++v) {
        float pv = __expf(sc[j][v] - mrun[v]);
        psum[v] += pv;
        Pw[(mb + v) * 64 + j * 16 + lr] = f2bf(pv);
      }
#pragma unroll
    for (int off = 8; off >= 1; off >>= 1)
#pragma unroll
      for (int v = 0; v < 8; ++v) psum[v] += __shfl_xor(psum[v], off, 32);
#pragma unroll
    for (int v = 0; v < 8; ++v) lrun[v] = lrun[v] * scl[v] + psum[v];
#pragma unroll
    for (int j = 0; j < 4; ++j)
#pragma unroll
      for (int v = 0; v < 8; ++v) ctx[j][v] *= scl[v];

    // ctx += P @ V (B from transposed V tile; same-wave P, DScnt ordered)
#pragma unroll
    for (int k0 = 0; k0 < 64; k0 += 32) {
      v16bf a = ld_frag(Pw, 64, lr, k0, kh);
      v16bf bfr[4];
#pragma unroll
      for (int j = 0; j < 4; ++j) bfr[j] = ld_frag(sVt, 64, j * 16 + lr, k0, kh);
#pragma unroll
      for (int j = 0; j < 4; ++j) ctx[j] = wmma_bf16(a, bfr[j], ctx[j]);
    }
  }

  // normalize and store to [b][l][h*64+d] row-major
#pragma unroll
  for (int v = 0; v < 8; ++v) lrun[v] = 1.0f / lrun[v];
#pragma unroll
  for (int j = 0; j < 4; ++j) {
    int d = j * 16 + lr;
#pragma unroll
    for (int v = 0; v < 8; ++v) {
      int l = l0 + wave * 16 + mb + v;
      Ctx[((size_t)b * S_DIM + l) * (NH * HD) + h * HD + d] =
          f2bf(ctx[j][v] * lrun[v]);
    }
  }
}

// ------------------------------------ output projection + bias + residual (f32)
__global__ __launch_bounds__(128) void out_gemm_kernel(
    const bf16_t* __restrict__ Cx, const bf16_t* __restrict__ W,
    const float* __restrict__ bo, const float* __restrict__ resid,
    float* __restrict__ H) {
  __shared__ bf16_t sX[64 * 64];
  __shared__ bf16_t sW[64 * 64];
  const int n0 = blockIdx.x * 64;
  const int m0 = blockIdx.y * 64;
  const int tid = threadIdx.x;
  const int wave = tid >> 5, lane = tid & 31;
  const int lr = lane & 15, kh = lane >> 4, mb = kh * 8;

  v8f acc[4];
#pragma unroll
  for (int j = 0; j < 4; ++j) acc[j] = vzero8();

#pragma unroll 1
  for (int k0 = 0; k0 < HID; k0 += 64) {
    __syncthreads();
#pragma unroll
    for (int it = 0; it < 4; ++it) {
      int idx = (it * 128 + tid) * 8;
      int r = idx >> 6, c = idx & 63;
      *(v8bf*)(sX + idx) = *(const v8bf*)(Cx + (size_t)(m0 + r) * HID + k0 + c);
      *(v8bf*)(sW + idx) = *(const v8bf*)(W + (size_t)(n0 + r) * HID + k0 + c);
    }
    __syncthreads();
    v16bf a0 = ld_frag(sX, 64, wave * 16 + lr, 0, kh);
    v16bf a1 = ld_frag(sX, 64, wave * 16 + lr, 32, kh);
    v16bf bfr[4];
#pragma unroll
    for (int j = 0; j < 4; ++j) bfr[j] = ld_frag(sW, 64, j * 16 + lr, 0, kh);
#pragma unroll
    for (int j = 0; j < 4; ++j) acc[j] = wmma_bf16(a0, bfr[j], acc[j]);
#pragma unroll
    for (int j = 0; j < 4; ++j) bfr[j] = ld_frag(sW, 64, j * 16 + lr, 32, kh);
#pragma unroll
    for (int j = 0; j < 4; ++j) acc[j] = wmma_bf16(a1, bfr[j], acc[j]);
  }

#pragma unroll
  for (int j = 0; j < 4; ++j) {
    int n = n0 + j * 16 + lr;
    float bsv = bo[n];
#pragma unroll
    for (int v = 0; v < 8; ++v) {
      int m = m0 + wave * 16 + mb + v;
      H[(size_t)m * HID + n] = acc[j][v] + bsv + resid[(size_t)m * HID + n];
    }
  }
}

// ---------------------------------------------------------- in-place LayerNorm
__global__ __launch_bounds__(256) void ln_kernel(float* __restrict__ H,
                                                 const float* __restrict__ g,
                                                 const float* __restrict__ bta) {
  const int row = blockIdx.x;
  float* p = H + (size_t)row * HID;
  const int tid = threadIdx.x;
  float x[4], s = 0.0f, q = 0.0f;
#pragma unroll
  for (int i = 0; i < 4; ++i) {
    x[i] = p[tid + i * 256];
    s += x[i];
    q += x[i] * x[i];
  }
#pragma unroll
  for (int off = 16; off >= 1; off >>= 1) {
    s += __shfl_xor(s, off, 32);
    q += __shfl_xor(q, off, 32);
  }
  __shared__ float ss[8], sq[8];
  if ((tid & 31) == 0) { ss[tid >> 5] = s; sq[tid >> 5] = q; }
  __syncthreads();
  float ts = 0.0f, tq = 0.0f;
#pragma unroll
  for (int w = 0; w < 8; ++w) { ts += ss[w]; tq += sq[w]; }
  const float mu = ts * (1.0f / HID);
  const float var = tq * (1.0f / HID) - mu * mu;
  const float rs = rsqrtf(var + 1e-12f);
#pragma unroll
  for (int i = 0; i < 4; ++i) {
    int c = tid + i * 256;
    p[c] = (x[i] - mu) * rs * g[c] + bta[c];
  }
}

extern "C" void kernel_launch(void* const* d_in, const int* in_sizes, int n_in,
                              void* d_out, int out_size, void* d_ws, size_t ws_size,
                              hipStream_t stream) {
  (void)in_sizes; (void)n_in; (void)out_size; (void)ws_size;
  const float* hs = (const float*)d_in[0];
  const float* am = (const float*)d_in[1];
  const float* Wq = (const float*)d_in[2];
  const float* bq = (const float*)d_in[3];
  const float* Wk = (const float*)d_in[4];
  const float* bk = (const float*)d_in[5];
  const float* Wv = (const float*)d_in[6];
  const float* bv = (const float*)d_in[7];
  const float* de = (const float*)d_in[8];
  const float* Wo = (const float*)d_in[9];
  const float* bo = (const float*)d_in[10];
  const float* lg = (const float*)d_in[11];
  const float* lb = (const float*)d_in[12];
  float* out = (float*)d_out;

  char* ws = (char*)d_ws;
  size_t off = 0;
  bf16_t* Xbf  = (bf16_t*)(ws + off); off += (size_t)4096 * 1024 * 2;
  bf16_t* Wqkv = (bf16_t*)(ws + off); off += (size_t)3072 * 1024 * 2;
  bf16_t* Wob  = (bf16_t*)(ws + off); off += (size_t)1024 * 1024 * 2;
  bf16_t* Ebf  = (bf16_t*)(ws + off); off += (size_t)2048 * 64 * 2;
  bf16_t* Qb   = (bf16_t*)(ws + off); off += (size_t)4096 * 1024 * 2;
  bf16_t* Kb   = (bf16_t*)(ws + off); off += (size_t)4096 * 1024 * 2;
  bf16_t* Vb   = (bf16_t*)(ws + off); off += (size_t)4096 * 1024 * 2;
  bf16_t* Cx   = (bf16_t*)(ws + off); off += (size_t)4096 * 1024 * 2;

  cvt_bf16_kernel<<<2048, 256, 0, stream>>>(hs, Xbf, 4096 * 1024, 4096 * 1024);
  cvt_bf16_kernel<<<512, 256, 0, stream>>>(Wq, Wqkv, 1024 * 1024, 1024 * 1024);
  cvt_bf16_kernel<<<512, 256, 0, stream>>>(Wk, Wqkv + 1024 * 1024, 1024 * 1024, 1024 * 1024);
  cvt_bf16_kernel<<<512, 256, 0, stream>>>(Wv, Wqkv + 2 * 1024 * 1024, 1024 * 1024, 1024 * 1024);
  cvt_bf16_kernel<<<512, 256, 0, stream>>>(Wo, Wob, 1024 * 1024, 1024 * 1024);
  cvt_bf16_kernel<<<64, 256, 0, stream>>>(de, Ebf, 2047 * 64, 2048 * 64);

  qkv_gemm_kernel<<<dim3(48, 64), 128, 0, stream>>>(Xbf, Wqkv, bq, bk, bv, Qb, Kb, Vb);
  attn_kernel<<<dim3(16, 16, 4), 128, 114688, stream>>>(Qb, Kb, Vb, Ebf, am, Cx);
  out_gemm_kernel<<<dim3(16, 64), 128, 0, stream>>>(Cx, Wob, bo, hs, out);
  ln_kernel<<<4096, 256, 0, stream>>>(out, lg, lb);
}
